// LiquidKANS6Cell_36747740184977
// MI455X (gfx1250) — matrix operand order repaired and checked by reading, not compiled, not run on hardware
//
#include <hip/hip_runtime.h>
#include <cstddef>

// ---------------------------------------------------------------------------
// Problem dims (compile-time constants from the reference)
// ---------------------------------------------------------------------------
#define DMODEL 768
#define DSTATE 16
#define NGRID  3
#define BATCH  2
#define SEQ    2048
#define ROWS   (BATCH * SEQ)           // 4096
#define NFEAT  (DMODEL * 4)            // 3072 : silu + 3 RBF per channel
#define NKAN   (DMODEL + 2 * DSTATE)   // 800  : [delta | B | C]
#define NXZ    (2 * DMODEL)            // 1536
#define NCHUNK 16
#define CHUNK  (SEQ / NCHUNK)          // 128

typedef __attribute__((ext_vector_type(16))) __bf16 bf16x16;
typedef __attribute__((ext_vector_type(8)))  __bf16 bf16x8;
typedef __attribute__((ext_vector_type(8)))  float  f32x8;

// ---------------------------------------------------------------------------
// Packing / conversion kernels (f32 -> bf16, transposed weight layouts)
// ---------------------------------------------------------------------------
__global__ void k_cast_bf16(const float* __restrict__ in, __bf16* __restrict__ out, int n) {
  int i = blockIdx.x * blockDim.x + threadIdx.x;
  if (i < n) out[i] = (__bf16)in[i];
}

// W [d][o] f32  ->  Wt [o][d] bf16   (pre-transpose so WMMA B operand is K-contiguous)
__global__ void k_pack_T(const float* __restrict__ W, __bf16* __restrict__ Wt, int d_dim, int o_dim) {
  int i = blockIdx.x * blockDim.x + threadIdx.x;
  if (i >= d_dim * o_dim) return;
  int o = i / d_dim, d = i % d_dim;
  Wt[(size_t)o * d_dim + d] = (__bf16)W[(size_t)d * o_dim + o];
}

// Fused KAN weight: WcatT [800][3072] bf16. Feature k = 4*d + c:
//   c==0 -> base weight (applied to silu(xn));  c==1..3 -> spline weight, grid g=c-1.
__global__ void k_pack_wcatT(const float* __restrict__ kd_b, const float* __restrict__ kd_s,
                             const float* __restrict__ kb_b, const float* __restrict__ kb_s,
                             const float* __restrict__ kc_b, const float* __restrict__ kc_s,
                             __bf16* __restrict__ WcatT) {
  int i = blockIdx.x * blockDim.x + threadIdx.x;
  if (i >= NKAN * NFEAT) return;
  int j = i / NFEAT;            // output column 0..799
  int k = i % NFEAT;            // feature index
  int d = k >> 2, c = k & 3;
  float v;
  if (j < DMODEL) {             // delta head, o = j
    v = (c == 0) ? kd_b[(size_t)d * DMODEL + j]
                 : kd_s[((size_t)d * NGRID + (c - 1)) * DMODEL + j];
  } else if (j < DMODEL + DSTATE) {  // B head
    int o = j - DMODEL;
    v = (c == 0) ? kb_b[(size_t)d * DSTATE + o]
                 : kb_s[((size_t)d * NGRID + (c - 1)) * DSTATE + o];
  } else {                      // C head
    int o = j - DMODEL - DSTATE;
    v = (c == 0) ? kc_b[(size_t)d * DSTATE + o]
                 : kc_s[((size_t)d * NGRID + (c - 1)) * DSTATE + o];
  }
  WcatT[i] = (__bf16)v;
}

// A = -exp(A_log)  and  pexp = exp(-priority)
__global__ void k_small_precomp(const float* __restrict__ A_log, float* __restrict__ Amat,
                                const float* __restrict__ prio, float* __restrict__ pexp) {
  int i = blockIdx.x * blockDim.x + threadIdx.x;
  if (i < DMODEL * DSTATE) Amat[i] = -__expf(A_log[i]);
  if (i < ROWS)            pexp[i] = __expf(-prio[i]);
}

// ---------------------------------------------------------------------------
// bf16 WMMA GEMM:  C[M,N] = A[M,K] * Bt[N,K]^T (+ bias)
// 2x2 register tiling: each wave owns a 32x32 output block (4 accumulators),
// so every A/B fragment load feeds two WMMAs. 4 waves/block -> 32x128 per block.
// ---------------------------------------------------------------------------
__global__ __launch_bounds__(128)
void k_gemm_wmma(const __bf16* __restrict__ A, const __bf16* __restrict__ Bt,
                 const float* __restrict__ bias, float* __restrict__ C0,
                 float* __restrict__ C1, int M, int N, int K, int splitN) {
  const int wave = threadIdx.x >> 5;
  const int lane = threadIdx.x & 31;
  const int half = lane >> 4;     // lane group 0/1
  const int r    = lane & 15;     // A: row-in-tile, B: col-in-tile
  const int tileN = (blockIdx.x * 4 + wave) * 32;
  const int tileM = blockIdx.y * 32;
  if (tileN >= N) return;

  const __bf16* Arow0 = A  + (size_t)(tileM + r)      * K;
  const __bf16* Arow1 = A  + (size_t)(tileM + 16 + r) * K;
  const __bf16* Brow0 = Bt + (size_t)(tileN + r)      * K;
  const __bf16* Brow1 = Bt + (size_t)(tileN + 16 + r) * K;

  f32x8 acc00 = {}, acc01 = {}, acc10 = {}, acc11 = {};
  for (int kk = 0; kk < K; kk += 32) {
    // A 16x32 bf16: lane-half 0 holds K {0..7,16..23}, half 1 holds K {8..15,24..31}
    bf16x8 a0l = *(const bf16x8*)(Arow0 + kk + half * 8);
    bf16x8 a0h = *(const bf16x8*)(Arow0 + kk + 16 + half * 8);
    bf16x8 a1l = *(const bf16x8*)(Arow1 + kk + half * 8);
    bf16x8 a1h = *(const bf16x8*)(Arow1 + kk + 16 + half * 8);
    // B 32x16 bf16 (stored transposed): lane-half 0 holds K {0..7,16..23}, half 1 holds K {8..15,24..31}
    bf16x8 b0l = *(const bf16x8*)(Brow0 + kk + half * 8);
    bf16x8 b0h = *(const bf16x8*)(Brow0 + kk + 16 + half * 8);
    bf16x8 b1l = *(const bf16x8*)(Brow1 + kk + half * 8);
    bf16x8 b1h = *(const bf16x8*)(Brow1 + kk + 16 + half * 8);
    bf16x16 a0, a1, b0, b1;
#pragma unroll
    for (int i = 0; i < 8; i++) {
      a0[i] = a0l[i]; a0[8 + i] = a0h[i];
      a1[i] = a1l[i]; a1[8 + i] = a1h[i];
      b0[i] = b0l[i]; b0[8 + i] = b0h[i];
      b1[i] = b1l[i]; b1[8 + i] = b1h[i];
    }
    acc00 = __builtin_amdgcn_wmma_f32_16x16x32_bf16(false, a0, false, b0, (short)0, acc00, false, false);
    acc01 = __builtin_amdgcn_wmma_f32_16x16x32_bf16(false, a0, false, b1, (short)0, acc01, false, false);
    acc10 = __builtin_amdgcn_wmma_f32_16x16x32_bf16(false, a1, false, b0, (short)0, acc10, false, false);
    acc11 = __builtin_amdgcn_wmma_f32_16x16x32_bf16(false, a1, false, b1, (short)0, acc11, false, false);
  }

  auto store_tile = [&](const f32x8& acc, int rowBase, int colBase) {
#pragma unroll
    for (int v = 0; v < 8; v++) {         // D: row = v + half*8, col = r
      int row = rowBase + half * 8 + v;
      int col = colBase + r;
      float val = acc[v] + (bias ? bias[col] : 0.f);
      if (C1) {                           // split mode (xz -> x_proj | z)
        if (col < splitN) C0[(size_t)row * splitN + col] = val;
        else              C1[(size_t)row * splitN + (col - splitN)] = val;
      } else {
        C0[(size_t)row * N + col] = val;
      }
    }
  };
  store_tile(acc00, tileM,      tileN);
  store_tile(acc01, tileM,      tileN + 16);
  store_tile(acc10, tileM + 16, tileN);
  store_tile(acc11, tileM + 16, tileN + 16);
}

// ---------------------------------------------------------------------------
// LayerNorm + KAN feature build: F[row][4d+{0..3}] = silu(xn), exp(-(xn-g)^2)
// ---------------------------------------------------------------------------
__global__ __launch_bounds__(256)
void k_features(const float* __restrict__ xproj, const float* __restrict__ ln_g,
                const float* __restrict__ ln_b, __bf16* __restrict__ F) {
  const int row = blockIdx.x;
  const float* xr = xproj + (size_t)row * DMODEL;
  float s = 0.f, s2 = 0.f;
  for (int d = threadIdx.x; d < DMODEL; d += 256) { float v = xr[d]; s += v; s2 += v * v; }
  for (int m = 16; m >= 1; m >>= 1) { s += __shfl_xor(s, m, 32); s2 += __shfl_xor(s2, m, 32); }
  __shared__ float ws[8], ws2[8];
  int wid = threadIdx.x >> 5, lane = threadIdx.x & 31;
  if (lane == 0) { ws[wid] = s; ws2[wid] = s2; }
  __syncthreads();
  if (threadIdx.x == 0) {
    float t = 0.f, t2 = 0.f;
    for (int i = 0; i < 8; i++) { t += ws[i]; t2 += ws2[i]; }
    float mu = t / DMODEL;
    ws[0]  = mu;
    ws2[0] = t2 / DMODEL - mu * mu;
  }
  __syncthreads();
  const float mu  = ws[0];
  const float inv = rsqrtf(ws2[0] + 1e-5f);
  for (int d = threadIdx.x; d < DMODEL; d += 256) {
    float xn = (xr[d] - mu) * inv * ln_g[d] + ln_b[d];
    float si = xn / (1.f + __expf(-xn));
    float dm = xn + 1.f, dz = xn, dp = xn - 1.f;   // grid = {-1, 0, +1}
    __bf16* Fr = F + (size_t)row * NFEAT + 4 * d;
    Fr[0] = (__bf16)si;
    Fr[1] = (__bf16)__expf(-dm * dm);
    Fr[2] = (__bf16)__expf(-dz * dz);
    Fr[3] = (__bf16)__expf(-dp * dp);
  }
}

// delta[row,d] = softplus(preact[row,d]) * exp(-priority[row])   (hoisted once)
__global__ void k_delta(const float* __restrict__ preact, const float* __restrict__ pexp,
                        float* __restrict__ delta) {
  int i = blockIdx.x * blockDim.x + threadIdx.x;
  if (i >= ROWS * DMODEL) return;
  int row = i / DMODEL, d = i % DMODEL;
  float v = preact[(size_t)row * NKAN + d];
  float sp = (v > 20.f) ? v : log1pf(__expf(v));
  delta[i] = sp * pexp[row];
}

// ---------------------------------------------------------------------------
// Chunked parallel scan (3 passes).  Lane layout per wave: 2 d-channels x 16 n.
// Pass 1: per-chunk local scan from h=0; store (prod dA, h_end).
// Pass 2: tiny sequential scan over the 16 chunk summaries per (b,d,n).
// Pass 3: re-run each chunk seeded with incoming state; emit y with C-reduce.
// ---------------------------------------------------------------------------
__global__ __launch_bounds__(32)
void k_scan_pass1(const float* __restrict__ delta, const float* __restrict__ preact,
                  const float* __restrict__ xproj, const float* __restrict__ Amat,
                  float* __restrict__ chunkA, float* __restrict__ chunkH) {
  const int blk  = blockIdx.x;                 // ((b*NCHUNK + c) * (D/2) + dp)
  const int dp   = blk % (DMODEL / 2);
  const int t    = blk / (DMODEL / 2);
  const int c    = t % NCHUNK;
  const int b    = t / NCHUNK;
  const int lane = threadIdx.x & 31;
  const int half = lane >> 4;
  const int n    = lane & 15;
  const int d    = dp * 2 + half;

  const float A = Amat[(size_t)d * DSTATE + n];
  float aP = 1.f, h = 0.f;
  const int l0 = c * CHUNK;
  for (int l = l0; l < l0 + CHUNK; l++) {
    const size_t row = (size_t)b * SEQ + l;
    if ((l & 15) == 0 && l + 16 < l0 + CHUNK)
      __builtin_prefetch(preact + (row + 16) * NKAN, 0, 1);   // global_prefetch_b8
    float dl = delta[row * DMODEL + d];
    float Bn = preact[row * NKAN + DMODEL + n];
    float xp = xproj[row * DMODEL + d];
    float dA = __expf(dl * A);
    h  = dA * h + dl * Bn * xp;
    aP *= dA;
  }
  size_t idx = (((size_t)b * NCHUNK + c) * DMODEL + d) * DSTATE + n;
  chunkA[idx] = aP;
  chunkH[idx] = h;
}

__global__ void k_chunk_combine(const float* __restrict__ chunkA, const float* __restrict__ chunkH,
                                float* __restrict__ chunkInit) {
  int i = blockIdx.x * blockDim.x + threadIdx.x;      // over B*DMODEL*DSTATE
  if (i >= BATCH * DMODEL * DSTATE) return;
  int b  = i / (DMODEL * DSTATE);
  int dn = i % (DMODEL * DSTATE);
  float carry = 0.f;
  for (int c = 0; c < NCHUNK; c++) {
    size_t idx = ((size_t)b * NCHUNK + c) * DMODEL * DSTATE + dn;
    chunkInit[idx] = carry;
    carry = chunkA[idx] * carry + chunkH[idx];
  }
}

__global__ __launch_bounds__(32)
void k_scan_pass3(const float* __restrict__ delta, const float* __restrict__ preact,
                  const float* __restrict__ xproj, const float* __restrict__ Amat,
                  const float* __restrict__ Dp, const float* __restrict__ chunkInit,
                  float* __restrict__ ycore) {
  const int blk  = blockIdx.x;
  const int dp   = blk % (DMODEL / 2);
  const int t    = blk / (DMODEL / 2);
  const int c    = t % NCHUNK;
  const int b    = t / NCHUNK;
  const int lane = threadIdx.x & 31;
  const int half = lane >> 4;
  const int n    = lane & 15;
  const int d    = dp * 2 + half;

  const float A  = Amat[(size_t)d * DSTATE + n];
  const float Dv = Dp[d];
  float h = chunkInit[(((size_t)b * NCHUNK + c) * DMODEL + d) * DSTATE + n];
  const int l0 = c * CHUNK;
  for (int l = l0; l < l0 + CHUNK; l++) {
    const size_t row = (size_t)b * SEQ + l;
    if ((l & 15) == 0 && l + 16 < l0 + CHUNK)
      __builtin_prefetch(preact + (row + 16) * NKAN, 0, 1);
    float dl = delta[row * DMODEL + d];
    float Bn = preact[row * NKAN + DMODEL + n];
    float Cn = preact[row * NKAN + DMODEL + DSTATE + n];
    float xp = xproj[row * DMODEL + d];
    float dA = __expf(dl * A);
    h = dA * h + dl * Bn * xp;
    float acc = Cn * h;
    acc += __shfl_xor(acc, 1, 32);   // reduce over n within each 16-lane group
    acc += __shfl_xor(acc, 2, 32);
    acc += __shfl_xor(acc, 4, 32);
    acc += __shfl_xor(acc, 8, 32);
    if (n == 0) ycore[row * DMODEL + d] = acc + Dv * xp;
  }
}

// y_gated = ycore * silu(z)  -> bf16 for the output GEMM
__global__ void k_gate(const float* __restrict__ ycore, const float* __restrict__ z,
                       __bf16* __restrict__ ybf, int nelem) {
  int i = blockIdx.x * blockDim.x + threadIdx.x;
  if (i >= nelem) return;
  float zv = z[i];
  ybf[i] = (__bf16)(ycore[i] * (zv / (1.f + __expf(-zv))));
}

// ---------------------------------------------------------------------------
// Host-side orchestration
// ---------------------------------------------------------------------------
extern "C" void kernel_launch(void* const* d_in, const int* in_sizes, int n_in,
                              void* d_out, int out_size, void* d_ws, size_t ws_size,
                              hipStream_t stream) {
  const float* x       = (const float*)d_in[0];
  const float* prio    = (const float*)d_in[1];
  const float* W_in    = (const float*)d_in[2];
  const float* b_in    = (const float*)d_in[3];
  const float* ln_g    = (const float*)d_in[4];
  const float* ln_b    = (const float*)d_in[5];
  const float* kd_b    = (const float*)d_in[6];
  const float* kd_s    = (const float*)d_in[7];
  const float* kb_b    = (const float*)d_in[8];
  const float* kb_s    = (const float*)d_in[9];
  const float* kc_b    = (const float*)d_in[10];
  const float* kc_s    = (const float*)d_in[11];
  const float* A_log   = (const float*)d_in[12];
  const float* D_param = (const float*)d_in[13];
  const float* W_out   = (const float*)d_in[14];
  const float* b_out   = (const float*)d_in[15];
  float* out = (float*)d_out;
  (void)in_sizes; (void)n_in; (void)out_size; (void)ws_size;

  // --- workspace carve-up (256B aligned) ---
  char* w = (char*)d_ws;
  auto carve = [&](size_t bytes) -> void* {
    void* p = (void*)w;
    w += (bytes + 255) & ~(size_t)255;
    return p;
  };
  __bf16* xbf     = (__bf16*)carve((size_t)ROWS * DMODEL * 2);
  __bf16* WinT    = (__bf16*)carve((size_t)NXZ * DMODEL * 2);
  float*  xproj   = (float*) carve((size_t)ROWS * DMODEL * 4);
  float*  zbuf    = (float*) carve((size_t)ROWS * DMODEL * 4);
  __bf16* F       = (__bf16*)carve((size_t)ROWS * NFEAT * 2);
  __bf16* WcatT   = (__bf16*)carve((size_t)NKAN * NFEAT * 2);
  float*  preact  = (float*) carve((size_t)ROWS * NKAN * 4);
  float*  Amat    = (float*) carve((size_t)DMODEL * DSTATE * 4);
  float*  pexp    = (float*) carve((size_t)ROWS * 4);
  float*  deltab  = (float*) carve((size_t)ROWS * DMODEL * 4);
  float*  chunkA  = (float*) carve((size_t)BATCH * NCHUNK * DMODEL * DSTATE * 4);
  float*  chunkH  = (float*) carve((size_t)BATCH * NCHUNK * DMODEL * DSTATE * 4);
  float*  chunkI  = (float*) carve((size_t)BATCH * NCHUNK * DMODEL * DSTATE * 4);
  float*  ycore   = (float*) carve((size_t)ROWS * DMODEL * 4);
  __bf16* ybf     = (__bf16*)carve((size_t)ROWS * DMODEL * 2);
  __bf16* WoutT   = (__bf16*)carve((size_t)DMODEL * DMODEL * 2);

  const int T = 256;
  // 0) pack / convert
  k_cast_bf16<<<(ROWS * DMODEL + T - 1) / T, T, 0, stream>>>(x, xbf, ROWS * DMODEL);
  k_pack_T<<<(DMODEL * NXZ + T - 1) / T, T, 0, stream>>>(W_in, WinT, DMODEL, NXZ);
  k_pack_wcatT<<<(NKAN * NFEAT + T - 1) / T, T, 0, stream>>>(kd_b, kd_s, kb_b, kb_s, kc_b, kc_s, WcatT);
  k_pack_T<<<(DMODEL * DMODEL + T - 1) / T, T, 0, stream>>>(W_out, WoutT, DMODEL, DMODEL);
  k_small_precomp<<<(ROWS + T - 1) / T, T, 0, stream>>>(A_log, Amat, prio, pexp);

  // 1) xz = x @ W_in + b_in  (split into x_proj | z)
  {
    dim3 grid((NXZ / 32 + 3) / 4, ROWS / 32);
    k_gemm_wmma<<<grid, 128, 0, stream>>>(xbf, WinT, b_in, xproj, zbuf,
                                          ROWS, NXZ, DMODEL, DMODEL);
  }
  // 2) LayerNorm + KAN features
  k_features<<<ROWS, 256, 0, stream>>>(xproj, ln_g, ln_b, F);

  // 3) fused KAN dense: preact[rows, 800] = F @ Wcat
  {
    dim3 grid((NKAN / 32 + 3) / 4, ROWS / 32);
    k_gemm_wmma<<<grid, 128, 0, stream>>>(F, WcatT, nullptr, preact, nullptr,
                                          ROWS, NKAN, NFEAT, 0);
  }
  // 4) delta = softplus(.) * exp(-priority)   (hoisted out of the scan loops)
  k_delta<<<(ROWS * DMODEL + T - 1) / T, T, 0, stream>>>(preact, pexp, deltab);

  // 5) chunked parallel scan: local scans -> chunk combine -> seeded re-scan + C-reduce
  {
    const int nblk = BATCH * NCHUNK * (DMODEL / 2);   // 12288 waves
    k_scan_pass1<<<nblk, 32, 0, stream>>>(deltab, preact, xproj, Amat, chunkA, chunkH);
    k_chunk_combine<<<(BATCH * DMODEL * DSTATE + T - 1) / T, T, 0, stream>>>(chunkA, chunkH, chunkI);
    k_scan_pass3<<<nblk, 32, 0, stream>>>(deltab, preact, xproj, Amat, D_param, chunkI, ycore);
  }

  // 6) gate by silu(z)
  k_gate<<<(ROWS * DMODEL + T - 1) / T, T, 0, stream>>>(ycore, zbuf, ybf, ROWS * DMODEL);

  // 7) out = y @ W_out + b_out
  {
    dim3 grid((DMODEL / 32 + 3) / 4, ROWS / 32);
    k_gemm_wmma<<<grid, 128, 0, stream>>>(ybf, WoutT, b_out, out, nullptr,
                                          ROWS, DMODEL, DMODEL, 0);
  }
}
